// GenericRNNSequenceEncoder_8890582303460
// MI455X (gfx1250) — matrix operand order repaired and checked by reading, not compiled, not run on hardware
//
#include <hip/hip_runtime.h>
#include <hip/hip_bf16.h>

// Problem sizes (fixed by the reference)
#define BB 64
#define TT 512
#define DD 1024
#define HH 1024
#define GG (4 * HH)   // 4096 gate columns
#define KK (DD + HH)  // 2048 fused K

typedef __attribute__((ext_vector_type(16))) __bf16 v16bf;
typedef __attribute__((ext_vector_type(8)))  float  v8f;

union BFx16 { v16bf v; unsigned int u[8]; };

__device__ __forceinline__ unsigned int bf16_rne(float f) {
  unsigned int u = __float_as_uint(f);
  u = (u + 0x7FFFu + ((u >> 16) & 1u)) >> 16;
  return u & 0xFFFFu;
}

// One v_perm_b32: result = {hi[31:16], lo[31:16]} = packed (bf16(lo), bf16(hi))
// Truncating f32->bf16; matches the ISA A-operand pair packing (K even in [15:0]).
__device__ __forceinline__ unsigned int pack_bf16_perm(float lo, float hi) {
  return __builtin_amdgcn_perm(__float_as_uint(hi), __float_as_uint(lo),
                               0x07060302u);
}

// Branch-free activations: v_exp_f32 + v_rcp_f32 only.
__device__ __forceinline__ float fast_sigmoid(float x) {
  return __builtin_amdgcn_rcpf(1.0f + __expf(-x));  // x->+inf: 1, x->-inf: 0
}
__device__ __forceinline__ float fast_tanh(float x) {
  // 2*sigmoid(2x) - 1; saturates correctly at both infinities.
  return fmaf(2.0f, __builtin_amdgcn_rcpf(1.0f + __expf(-2.0f * x)), -1.0f);
}

// ---------------------------------------------------------------------------
// Prep: Wt[n][k] (bf16, row-major over K=2048) = concat(Wi, Wh) transposed.
// Row-major-over-K means a lane's (k, k+1) bf16 pair is one contiguous u32,
// exactly the ISA B-operand packing for V_WMMA_*_16X16X32_BF16.
// ---------------------------------------------------------------------------
__global__ void lstm_prep_weights(const float* __restrict__ Wi,
                                  const float* __restrict__ Wh,
                                  unsigned short* __restrict__ Wt) {
  size_t tid = (size_t)blockIdx.x * blockDim.x + threadIdx.x;
  if (tid >= (size_t)GG * KK) return;
  int n = (int)(tid / KK);
  int k = (int)(tid % KK);
  float w = (k < DD) ? Wi[(size_t)k * GG + n] : Wh[(size_t)(k - DD) * GG + n];
  Wt[tid] = (unsigned short)bf16_rne(w);
}

// ---------------------------------------------------------------------------
// One LSTM timestep. 256 waves total; wave w owns the 16x16 patch
// (m_tile = w&3, n_tile = w>>2) of all four gates.
//   z = [x_t ; h_prev] @ Wt^T   (bf16 WMMA, fp32 accum, K = 2048)
// ---------------------------------------------------------------------------
__global__ void lstm_step(const float* __restrict__ x,      // [B,T,D]
                          const unsigned short* __restrict__ Wt, // [G,K] bf16
                          const float* __restrict__ bvec,   // [4H]
                          const int*   __restrict__ lengths,// [B]
                          const float* __restrict__ hprev,  // [B,H]
                          float* __restrict__ hnext,        // [B,H]
                          float* __restrict__ cbuf,         // [B,H] in/out
                          float* __restrict__ out,          // [B,T,H]
                          float* __restrict__ fc,           // [B,H]
                          float* __restrict__ fh,           // [B,H]
                          int t) {
  const int lane = threadIdx.x & 31;
  const int wid  = (blockIdx.x * (blockDim.x >> 5)) + (threadIdx.x >> 5);
  const int m_tile = wid & 3;    // 4 tiles of 16 over B=64
  const int n_tile = wid >> 2;   // 64 tiles of 16 over H=1024

  const int half = lane >> 4;        // 0: lanes 0-15, 1: lanes 16-31
  const int l15  = lane & 15;
  const int mA   = m_tile * 16 + l15;      // A-operand row for this lane
  const int ncol = n_tile * 16 + l15;      // B/C/D column for this lane

  v8f acc[4] = {};   // i, f, g, o accumulators (fp32)

  const float* xrow = x + ((size_t)mA * TT + t) * DD;
  const float* hrow = hprev + (size_t)mA * HH;

  // Pull next timestep's x-row toward the caches while this step computes
  // (emits global_prefetch_b8; speculative, safe).
  {
    int tn = (t + 1 < TT) ? (t + 1) : t;
    __builtin_prefetch(x + ((size_t)mA * TT + tn) * DD + (half << 6), 0, 3);
  }

  for (int k0 = 0; k0 < KK; k0 += 32) {
    // ---- A operand: 16x32 bf16 tile of [x_t ; h] for rows m_tile*16..+15
    const float* arow = (k0 < DD) ? (xrow + k0) : (hrow + (k0 - DD));
    BFx16 A;
#pragma unroll
    for (int p = 0; p < 8; ++p) {
      // ISA A layout: vgpr p holds K = (p>=4?16:0) + half*8 + (p&3)*2 + {0,1}
      int kk = ((p >> 2) << 4) + (half << 3) + ((p & 3) << 1);
      A.u[p] = pack_bf16_perm(arow[kk], arow[kk + 1]);
    }
    // ---- 4 gates share A; B rows are contiguous bf16 pairs in Wt
#pragma unroll
    for (int g = 0; g < 4; ++g) {
      const unsigned int* wrow =
          (const unsigned int*)(Wt + (size_t)(g * HH + ncol) * KK) + (k0 >> 1);
      BFx16 Bm;
#pragma unroll
      for (int p = 0; p < 8; ++p) {
        // B layout matches A: vgpr p holds K = (p>=4?16:0) + half*8 + (p&3)*2 + {0,1}
        Bm.u[p] = wrow[((p >> 2) << 3) + (half << 2) + (p & 3)];
      }
      acc[g] = __builtin_amdgcn_wmma_f32_16x16x32_bf16(
          false, A.v, false, Bm.v, (short)0, acc[g], false, false);
    }
  }

  // ---- gate math + state update; C/D layout: vgpr r -> row r + 8*half
  float bias0 = bvec[0 * HH + ncol];
  float bias1 = bvec[1 * HH + ncol];
  float bias2 = bvec[2 * HH + ncol];
  float bias3 = bvec[3 * HH + ncol];

#pragma unroll
  for (int r = 0; r < 8; ++r) {
    int mrow = m_tile * 16 + r + (half << 3);
    float iv = fast_sigmoid(acc[0][r] + bias0);
    float fv = fast_sigmoid(acc[1][r] + bias1);
    float gv = fast_tanh   (acc[2][r] + bias2);
    float ov = fast_sigmoid(acc[3][r] + bias3);

    size_t hidx = (size_t)mrow * HH + ncol;
    float cold = cbuf[hidx];
    float cnew = fv * cold + iv * gv;
    float hnew = ov * fast_tanh(cnew);

    cbuf[hidx]  = cnew;
    hnext[hidx] = hnew;
    out[((size_t)mrow * TT + t) * HH + ncol] = hnew;
    if (t == lengths[mrow] - 1) { fc[hidx] = cnew; fh[hidx] = hnew; }
  }
}

// ---------------------------------------------------------------------------
extern "C" void kernel_launch(void* const* d_in, const int* in_sizes, int n_in,
                              void* d_out, int out_size, void* d_ws, size_t ws_size,
                              hipStream_t stream) {
  const float* x       = (const float*)d_in[0];  // [B,T,D]
  const int*   lengths = (const int*)  d_in[1];  // [B]
  const float* init_c  = (const float*)d_in[2];  // [B,H]
  const float* init_h  = (const float*)d_in[3];  // [B,H]
  const float* Wi      = (const float*)d_in[4];  // [D,4H]
  const float* Wh      = (const float*)d_in[5];  // [H,4H]
  const float* bvec    = (const float*)d_in[6];  // [4H]

  float* out = (float*)d_out;                          // [B,T,H]
  float* fc  = out + (size_t)BB * TT * HH;             // [B,H]
  float* fh  = fc + (size_t)BB * HH;                   // [B,H]

  // Workspace: Wt bf16 (16 MB) | hbuf0 | hbuf1 | cbuf (256 KB each)
  unsigned short* Wt = (unsigned short*)d_ws;
  float* hbuf0 = (float*)((char*)d_ws + (size_t)GG * KK * sizeof(unsigned short));
  float* hbuf1 = hbuf0 + (size_t)BB * HH;
  float* cbuf  = hbuf1 + (size_t)BB * HH;

  // Init recurrent state
  hipMemcpyAsync(hbuf0, init_h, (size_t)BB * HH * sizeof(float),
                 hipMemcpyDeviceToDevice, stream);
  hipMemcpyAsync(cbuf, init_c, (size_t)BB * HH * sizeof(float),
                 hipMemcpyDeviceToDevice, stream);

  // Weight transpose + bf16 conversion (one pass, L2-resident thereafter)
  {
    size_t total = (size_t)GG * KK;
    int blk = 256;
    int grd = (int)((total + blk - 1) / blk);
    lstm_prep_weights<<<grd, blk, 0, stream>>>(Wi, Wh, Wt);
  }

  // Sequential scan: one launch per timestep, h double-buffered
  for (int t = 0; t < TT; ++t) {
    float* hr = (t & 1) ? hbuf1 : hbuf0;
    float* hw = (t & 1) ? hbuf0 : hbuf1;
    // 256 waves = 32 blocks x 8 waves of 32 (wave32)
    lstm_step<<<32, 256, 0, stream>>>(x, Wt, bvec, lengths, hr, hw, cbuf,
                                      out, fc, fh, t);
  }
}
